// GATEncoder_36962488549651
// MI455X (gfx1250) — matrix operand order repaired and checked by reading, not compiled, not run on hardware
//
#include <hip/hip_runtime.h>
#include <cstdint>
#include <cstddef>

// ---------------------------------------------------------------------------
// Types for WMMA fragments (gfx1250, wave32)
// ---------------------------------------------------------------------------
typedef __attribute__((ext_vector_type(16))) __bf16        v16bf;
typedef __attribute__((ext_vector_type(8)))  float         v8f;
typedef __attribute__((ext_vector_type(4)))  float         f32x4;
typedef __attribute__((ext_vector_type(4)))  unsigned int  u32x4;

#define BM 128
#define BN 64
#define BK 32

// ---------------------------------------------------------------------------
// Utility kernels
// ---------------------------------------------------------------------------
__global__ __launch_bounds__(256) void zero_f32_k(float* __restrict__ p, int n) {
  int i = blockIdx.x * 256 + threadIdx.x;
  if (i < n) p[i] = 0.f;
}

__global__ __launch_bounds__(256) void init_ms_k(float* __restrict__ m,
                                                 float* __restrict__ s, int n) {
  int i = blockIdx.x * 256 + threadIdx.x;
  if (i < n) { m[i] = -__builtin_inff(); s[i] = 0.f; }
}

// ---------------------------------------------------------------------------
// Tile staging: fp32 global -> bf16 LDS.  A: [BM][BK] row-major (zero-padded
// past M); B: [BN][BK] i.e. transposed so WMMA B fragments are contiguous.
// ---------------------------------------------------------------------------
__device__ __forceinline__ void stage_tiles(
    const float* __restrict__ A, const float* __restrict__ B,
    __bf16* __restrict__ As, __bf16* __restrict__ Bs,
    int m0, int n0, int kt, int M, int K, int N, int tid) {
  const int ar = tid >> 2;             // 0..63  (A row, two halves)
  const int ak = (tid & 3) * 8;        // 0..24  (A k chunk)
#pragma unroll
  for (int half = 0; half < 2; ++half) {
    int r  = ar + half * 64;
    int gr = m0 + r;
    union { __bf16 h[8]; u32x4 u; } pk;
    if (gr < M) {
      const float* ap = A + (size_t)gr * K + kt + ak;
      f32x4 a0 = *(const f32x4*)ap;
      f32x4 a1 = *(const f32x4*)(ap + 4);
#pragma unroll
      for (int j = 0; j < 4; ++j) pk.h[j]     = (__bf16)a0[j];
#pragma unroll
      for (int j = 0; j < 4; ++j) pk.h[4 + j] = (__bf16)a1[j];
      if (kt + BK < K) __builtin_prefetch(ap + BK, 0, 1);     // global_prefetch
    } else {
#pragma unroll
      for (int j = 0; j < 8; ++j) pk.h[j] = (__bf16)0.f;
    }
    *(u32x4*)(&As[r * BK + ak]) = pk.u;                       // ds_store_b128
  }
  const int bk = tid >> 3;             // 0..31  (B k row)
  const int bn = (tid & 7) * 8;        // 0..56  (B n chunk)
  const float* bp = B + (size_t)(kt + bk) * N + n0 + bn;
  f32x4 b0 = *(const f32x4*)bp;
  f32x4 b1 = *(const f32x4*)(bp + 4);
#pragma unroll
  for (int j = 0; j < 4; ++j) Bs[(bn + j) * BK + bk]     = (__bf16)b0[j];
#pragma unroll
  for (int j = 0; j < 4; ++j) Bs[(bn + 4 + j) * BK + bk] = (__bf16)b1[j];
  if (kt + BK < K) __builtin_prefetch(bp + (size_t)BK * N, 0, 1);
}

// ---------------------------------------------------------------------------
// GEMM: C[M,N] = A[M,K] (fp32) x B[K,N] (fp32), bf16 multiply / f32 accumulate
// 256-thread block (8 waves) computes a 128x64 tile; wave w owns rows
// [w*16,+16) and all 64 cols -> 4 WMMAs/k-step sharing one A fragment.
// Double-buffered LDS: one barrier per k-step; staging of tile k+1 overlaps
// the WMMA pipeline of tile k.
// ---------------------------------------------------------------------------
__global__ __launch_bounds__(256) void gemm_bf16_wmma_k(
    const float* __restrict__ A, const float* __restrict__ B,
    float* __restrict__ Cmat, int M, int K, int N) {
  __shared__ __bf16 As[2][BM * BK];    // [row][k]
  __shared__ __bf16 Bs[2][BN * BK];    // transposed: [n][k]

  const int tid  = threadIdx.x;
  const int lane = tid & 31;
  const int wv   = tid >> 5;           // 0..7
  const int m0   = blockIdx.y * BM;
  const int n0   = blockIdx.x * BN;

  v8f acc[4] = {};

  stage_tiles(A, B, As[0], Bs[0], m0, n0, 0, M, K, N, tid);
  int buf = 0;
  for (int kt = 0; kt < K; kt += BK) {
    __syncthreads();                             // staging of `buf` complete
    if (kt + BK < K)
      stage_tiles(A, B, As[buf ^ 1], Bs[buf ^ 1], m0, n0, kt + BK, M, K, N, tid);

    // ---- fragment loads (ISA 7.12.2 layouts) ----
    union { v16bf v; u32x4 u[2]; } af;
    {
      // A 16x32 bf16: lane<16 -> K 0..7 & 16..23; lane>=16 -> K 8..15 & 24..31
      int row = wv * 16 + (lane & 15);
      int kb  = (lane >> 4) * 8;
      af.u[0] = *(const u32x4*)(&As[buf][row * BK + kb]);
      af.u[1] = *(const u32x4*)(&As[buf][row * BK + kb + 16]);
    }
    const int kb2 = (lane >> 4) * 8;
#pragma unroll
    for (int t = 0; t < 4; ++t) {
      // B 32x16 bf16: lane = N col; lane<16 -> K 0..7 & 16..23; lane>=16 -> K 8..15 & 24..31
      union { v16bf v; u32x4 u[2]; } bf;
      int c = t * 16 + (lane & 15);
      bf.u[0] = *(const u32x4*)(&Bs[buf][c * BK + kb2]);
      bf.u[1] = *(const u32x4*)(&Bs[buf][c * BK + kb2 + 16]);
      acc[t] = __builtin_amdgcn_wmma_f32_16x16x32_bf16(false, af.v, false, bf.v,
                                                       (short)0, acc[t], false, false);
    }
    buf ^= 1;
  }

  // ---- store D: lane half selects M 0..7 vs 8..15, VGPR i = M offset ----
  int mrow = m0 + wv * 16 + (lane >> 4) * 8;
#pragma unroll
  for (int t = 0; t < 4; ++t) {
    int ncol = n0 + t * 16 + (lane & 15);
#pragma unroll
    for (int i = 0; i < 8; ++i) {
      int mr = mrow + i;
      if (mr < M) Cmat[(size_t)mr * N + ncol] = acc[t][i];
    }
  }
}

// ---------------------------------------------------------------------------
// a_s[n,h] = sum_c hs[n,h,c] * att[h,c]
// ---------------------------------------------------------------------------
__global__ __launch_bounds__(256) void reduce_att_k(
    const float* __restrict__ h, const float* __restrict__ att,
    float* __restrict__ out, int N, int H, int C) {
  int idx = blockIdx.x * 256 + threadIdx.x;
  if (idx >= N * H) return;
  int n = idx / H, hh = idx - n * H;
  const f32x4* hp = (const f32x4*)(h + (size_t)n * H * C + (size_t)hh * C);
  const f32x4* ap = (const f32x4*)(att + (size_t)hh * C);
  float s = 0.f;
  for (int c = 0; c < C / 4; ++c) {
    f32x4 x = hp[c], y = ap[c];
    s += x[0]*y[0] + x[1]*y[1] + x[2]*y[2] + x[3]*y[3];
  }
  out[idx] = s;
}

// Wv[k,h] = sum_c W_dst[k, h*C+c] * att_dst[h,c]   (tiny reduced weight)
__global__ __launch_bounds__(256) void wv_k(
    const float* __restrict__ W, const float* __restrict__ att,
    float* __restrict__ wv, int K, int H, int C) {
  int idx = blockIdx.x * 256 + threadIdx.x;
  if (idx >= K * H) return;
  int k = idx / H, hh = idx - k * H;
  const float* wp = W + (size_t)k * H * C + (size_t)hh * C;
  const float* ap = att + (size_t)hh * C;
  float s = 0.f;
  for (int c = 0; c < C; ++c) s += wp[c] * ap[c];
  wv[idx] = s;
}

// a_d[n,h] = x[n,:] . Wv[:,h]   — one wave per node, lanes partition K
__global__ __launch_bounds__(256) void matvec_ad_k(
    const float* __restrict__ X, const float* __restrict__ wv,
    float* __restrict__ ad, int N, int K, int H) {
  int lane = threadIdx.x & 31;
  int n = blockIdx.x * 8 + (threadIdx.x >> 5);
  if (n >= N) return;                       // uniform per wave
  float acch[2] = {0.f, 0.f};
  const float* xp = X + (size_t)n * K;
  for (int k = lane; k < K; k += 32) {
    float xv = xp[k];
    acch[0] += xv * wv[k * H + 0];
    if (H > 1) acch[1] += xv * wv[k * H + 1];
  }
  for (int hh = 0; hh < H; ++hh) {
    float v = acch[hh];
    for (int off = 16; off; off >>= 1) v += __shfl_xor(v, off, 32);
    if (lane == 0) ad[(size_t)n * H + hh] = v;
  }
}

// ---------------------------------------------------------------------------
// Edge phases: logits+segment-max, exp+segment-sum, weighted scatter-add
// ---------------------------------------------------------------------------
__device__ inline void atomicMaxF(float* addr, float v) {
  if (v >= 0.f) atomicMax((int*)addr, __float_as_int(v));
  else          atomicMin((unsigned int*)addr, __float_as_uint(v));
}

__global__ __launch_bounds__(256) void edge_logits_k(
    const int* __restrict__ src, const int* __restrict__ dst,
    const float* __restrict__ as_, const float* __restrict__ ad_,
    float* __restrict__ ebuf, float* __restrict__ mx, int E, int H) {
  int e = blockIdx.x * 256 + threadIdx.x;
  if (e >= E) return;
  int s_ = src[e], d_ = dst[e];
  for (int hh = 0; hh < H; ++hh) {
    float v = as_[(size_t)s_ * H + hh] + ad_[(size_t)d_ * H + hh];
    v = v > 0.f ? v : 0.2f * v;                       // leaky_relu
    ebuf[(size_t)e * H + hh] = v;
    atomicMaxF(&mx[(size_t)d_ * H + hh], v);
  }
}

__global__ __launch_bounds__(256) void edge_exp_k(
    const int* __restrict__ dst, float* __restrict__ ebuf,
    const float* __restrict__ mx, float* __restrict__ ssum, int E, int H) {
  int e = blockIdx.x * 256 + threadIdx.x;
  if (e >= E) return;
  int d_ = dst[e];
  for (int hh = 0; hh < H; ++hh) {
    float w = __expf(ebuf[(size_t)e * H + hh] - mx[(size_t)d_ * H + hh]);
    ebuf[(size_t)e * H + hh] = w;
    atomicAdd(&ssum[(size_t)d_ * H + hh], w);
  }
}

// one thread per (edge, 4-channel group): acc[dst] += hs[src] * alpha
__global__ __launch_bounds__(256) void edge_aggr_k(
    const int* __restrict__ src, const int* __restrict__ dst,
    const float* __restrict__ hs, const float* __restrict__ w,
    const float* __restrict__ ssum, float* __restrict__ accb,
    int E, int H, int C) {
  int HCq = H * C / 4;
  long long idx = (long long)blockIdx.x * 256 + threadIdx.x;
  if (idx >= (long long)E * HCq) return;
  int e  = (int)(idx / HCq);
  int g  = (int)(idx - (long long)e * HCq);
  int ch = g * 4;
  int hh = ch / C;
  int s_ = src[e], d_ = dst[e];
  float alpha = w[(size_t)e * H + hh] / (ssum[(size_t)d_ * H + hh] + 1e-16f);
  f32x4 hv = *(const f32x4*)(hs + (size_t)s_ * H * C + ch);
  float* ap = accb + (size_t)d_ * H * C + ch;
  atomicAdd(ap + 0, hv[0] * alpha);
  atomicAdd(ap + 1, hv[1] * alpha);
  atomicAdd(ap + 2, hv[2] * alpha);
  atomicAdd(ap + 3, hv[3] * alpha);
}

// bias (up to 3 relations per dst type) + optional ReLU; layer2 has H=1 so
// the concat=False head-mean is identity.
__global__ __launch_bounds__(256) void finalize_k(
    const float* __restrict__ accb, const float* b0, const float* b1,
    const float* b2, float* __restrict__ outp, int total, int HC, int doRelu) {
  int i = blockIdx.x * 256 + threadIdx.x;
  if (i >= total) return;
  int j = i % HC;
  float v = accb[i] + b0[j];
  if (b1) v += b1[j];
  if (b2) v += b2[j];
  if (doRelu) v = v > 0.f ? v : 0.f;
  outp[i] = v;
}

// ---------------------------------------------------------------------------
// Host orchestration
// ---------------------------------------------------------------------------
extern "C" void kernel_launch(void* const* d_in, const int* in_sizes, int n_in,
                              void* d_out, int out_size, void* d_ws, size_t ws_size,
                              hipStream_t stream) {
  (void)in_sizes; (void)n_in; (void)out_size; (void)ws_size;

  const int NN[4]    = {50000, 500, 100000, 20000};      // movie, genre, user, conv
  const int FEATN[4] = {256, 64, 128, 256};

  struct RelInfo { int st, dt, E, ein, ps; };
  // ps = alphabetical slot in params: has_genre(0), has_movie(1),
  // mentioned_in(2), mentions(3), rated_by(4), rated_high(5)
  const RelInfo rels[6] = {
    {0, 1, 150000,  4, 0},   // has_genre
    {1, 0, 150000,  6, 1},   // has_movie
    {2, 0, 400000,  8, 5},   // rated_high
    {0, 2, 400000, 10, 4},   // rated_by
    {3, 0, 200000, 12, 3},   // mentions
    {0, 3, 200000, 14, 2},   // mentioned_in
  };

  // workspace layout (floats)
  float* ws = (float*)d_ws;
  size_t o = 0;
  float* accB[4]; float* xB[4];
  for (int t = 0; t < 4; ++t) { accB[t] = ws + o; o += (size_t)NN[t] * 512; }
  for (int t = 0; t < 4; ++t) { xB[t]   = ws + o; o += (size_t)NN[t] * 512; }
  float* hsB = ws + o; o += (size_t)100000 * 512;
  float* asB = ws + o; o += 200000;
  float* adB = ws + o; o += 200000;
  float* mB  = ws + o; o += 200000;
  float* sB  = ws + o; o += 200000;
  float* eB  = ws + o; o += 800000;
  float* wvB = ws + o; o += 2048;

  // params flattened: layer ("0","1","2") -> relation (alpha) ->
  //   [W_dst, W_src, att_dst, att_src, bias]
  auto leaf = [&](int l, int ps, int f) -> const float* {
    return (const float*)d_in[16 + l * 30 + ps * 5 + f];
  };

  const int LH[3] = {2, 2, 1}, LC[3] = {256, 256, 384};
  size_t outOff[4];
  outOff[0] = 0;
  outOff[1] = outOff[0] + (size_t)50000  * 384;
  outOff[2] = outOff[1] + (size_t)500    * 384;
  outOff[3] = outOff[2] + (size_t)100000 * 384;

  for (int l = 0; l < 3; ++l) {
    const int H = LH[l], C = LC[l], HC = H * C;
    const float* Xin[4]; int Kin[4];
    for (int t = 0; t < 4; ++t) {
      Xin[t] = (l == 0) ? (const float*)d_in[t] : xB[t];
      Kin[t] = (l == 0) ? FEATN[t] : 512;
    }
    for (int t = 0; t < 4; ++t) {
      int n = NN[t] * HC;
      zero_f32_k<<<(n + 255) / 256, 256, 0, stream>>>(accB[t], n);
    }
    for (int r = 0; r < 6; ++r) {
      const RelInfo& R = rels[r];
      const float* Wd   = leaf(l, R.ps, 0);
      const float* Wsrc = leaf(l, R.ps, 1);
      const float* attD = leaf(l, R.ps, 2);
      const float* attS = leaf(l, R.ps, 3);
      const int* esrc = (const int*)d_in[R.ein];
      const int* edst = (const int*)d_in[R.ein + 1];
      const int Ns = NN[R.st], Nd = NN[R.dt];
      const int Ks = Kin[R.st], Kd = Kin[R.dt];

      // a_d without the full hd GEMM: Wv = W_dst . att_dst, then mat-vec
      wv_k<<<(Kd * H + 255) / 256, 256, 0, stream>>>(Wd, attD, wvB, Kd, H, C);
      matvec_ad_k<<<(Nd + 7) / 8, 256, 0, stream>>>(Xin[R.dt], wvB, adB, Nd, Kd, H);

      // hs = x_src @ W_src   (WMMA bf16 path)
      dim3 g(HC / 64, (Ns + 127) / 128);
      gemm_bf16_wmma_k<<<g, 256, 0, stream>>>(Xin[R.st], Wsrc, hsB, Ns, Ks, HC);

      reduce_att_k<<<(Ns * H + 255) / 256, 256, 0, stream>>>(hsB, attS, asB, Ns, H, C);
      init_ms_k<<<(Nd * H + 255) / 256, 256, 0, stream>>>(mB, sB, Nd * H);
      edge_logits_k<<<(R.E + 255) / 256, 256, 0, stream>>>(esrc, edst, asB, adB, eB, mB, R.E, H);
      edge_exp_k<<<(R.E + 255) / 256, 256, 0, stream>>>(edst, eB, mB, sB, R.E, H);

      long long tot = (long long)R.E * (HC / 4);
      edge_aggr_k<<<(unsigned)((tot + 255) / 256), 256, 0, stream>>>(
          esrc, edst, hsB, eB, sB, accB[R.dt], R.E, H, C);
    }
    // finalize per destination type: sum of relation biases (+ReLU), or output
    for (int t = 0; t < 4; ++t) {
      const float *b0 = nullptr, *b1 = nullptr, *b2 = nullptr;
      int cnt = 0;
      for (int r = 0; r < 6; ++r) if (rels[r].dt == t) {
        const float* bp = leaf(l, rels[r].ps, 4);
        if (cnt == 0) b0 = bp; else if (cnt == 1) b1 = bp; else b2 = bp;
        ++cnt;
      }
      float* dstp = (l == 2) ? ((float*)d_out + outOff[t]) : xB[t];
      int total = NN[t] * HC;
      finalize_k<<<(total + 255) / 256, 256, 0, stream>>>(
          accB[t], b0, b1, b2, dstp, total, HC, (l < 2) ? 1 : 0);
    }
  }
}
